// GraphSAGEModel_18047452577826
// MI455X (gfx1250) — compile-verified, matrix operand order checked by reading, not run on hardware
//
#include <hip/hip_runtime.h>
#include <hip/hip_bf16.h>

#define HID 64

typedef __attribute__((ext_vector_type(16))) __bf16 v16bf;
typedef __attribute__((ext_vector_type(8)))  float  v8f;

__global__ void zero_f32(float* __restrict__ p, long long n) {
    long long i = (long long)blockIdx.x * blockDim.x + threadIdx.x;
    if (i < n) p[i] = 0.0f;
}

__global__ void count_edges(const int* __restrict__ dst, float* __restrict__ cnt, int E) {
    int e = blockIdx.x * blockDim.x + threadIdx.x;
    if (e < E) atomicAdd(&cnt[dst[e]], 1.0f);
}

// Pack W [K,64] f32 row-major into WMMA B-fragment order, bf16.
// Fragment index = (c0*4 + t)*32 + lane ; 16 bf16 elements contiguous per lane.
// lane<16: K rows kb..kb+7 and kb+16..kb+23 (kb = (lane>>4)*8), col = t*16 + (lane&15).
__global__ void pack_weight(const float* __restrict__ W, __bf16* __restrict__ Wp, int K) {
    int idx = blockIdx.x * blockDim.x + threadIdx.x;   // (c0*4 + t)*32 + lane
    int total = (K / 32) * 4 * 32;
    if (idx >= total) return;
    int lane = idx & 31;
    int t    = (idx >> 5) & 3;
    int c0   = idx >> 7;
    int col  = t * 16 + (lane & 15);
    int kb   = (lane >> 4) * 8;
    int k0   = c0 * 32;
    __bf16* o = Wp + (long long)idx * 16;
    #pragma unroll
    for (int j = 0; j < 8; ++j) {
        o[j]     = (__bf16)W[(k0 + kb + j)      * HID + col];
        o[8 + j] = (__bf16)W[(k0 + kb + 16 + j) * HID + col];
    }
}

// Dual-projection GEMM: YL = X*Wl, YR = X*Wr.
// X: [N,K] f32 row-major; Wlp/Wrp: bf16 weights pre-packed in B-fragment order.
// One wave per 16-row tile; v_wmma_f32_16x16x32_bf16, f32 accumulation.
__global__ void __launch_bounds__(256) gemm_dual(const float* __restrict__ X,
                                                 const __bf16* __restrict__ Wlp,
                                                 const __bf16* __restrict__ Wrp,
                                                 float* __restrict__ YL,
                                                 float* __restrict__ YR,
                                                 int N, int K) {
    int wave  = (int)((blockIdx.x * blockDim.x + threadIdx.x) >> 5);
    int lane  = threadIdx.x & 31;
    int mbase = wave * 16;
    if (mbase >= N) return;

    int arow = mbase + (lane & 15);          // A-fragment row for this lane
    if (arow >= N) arow = N - 1;             // tail guard (N is a multiple of 16 here)
    int kb   = (lane >> 4) * 8;              // lane<16 -> K 0..7/16..23; lane>=16 -> 8..15/24..31
    int col  = lane & 15;                    // C column within 16-wide tile

    v8f accL[4] = {};
    v8f accR[4] = {};

    int c0 = 0;
    for (int k0 = 0; k0 < K; k0 += 32, ++c0) {
        // A fragment: two 16B-aligned float4 pairs -> bf16
        const float4* p0 = (const float4*)(X + (long long)arow * K + k0 + kb);
        const float4* p1 = (const float4*)(X + (long long)arow * K + k0 + kb + 16);
        float4 x0 = p0[0], x1 = p0[1];
        float4 x2 = p1[0], x3 = p1[1];
        v16bf a;
        a[0]  = (__bf16)x0.x; a[1]  = (__bf16)x0.y; a[2]  = (__bf16)x0.z; a[3]  = (__bf16)x0.w;
        a[4]  = (__bf16)x1.x; a[5]  = (__bf16)x1.y; a[6]  = (__bf16)x1.z; a[7]  = (__bf16)x1.w;
        a[8]  = (__bf16)x2.x; a[9]  = (__bf16)x2.y; a[10] = (__bf16)x2.z; a[11] = (__bf16)x2.w;
        a[12] = (__bf16)x3.x; a[13] = (__bf16)x3.y; a[14] = (__bf16)x3.z; a[15] = (__bf16)x3.w;

        #pragma unroll
        for (int t = 0; t < 4; ++t) {
            long long fo = (((long long)c0 * 4 + t) * 32 + lane) * 16;
            v16bf bl = *(const v16bf*)(Wlp + fo);
            v16bf br = *(const v16bf*)(Wrp + fo);
            accL[t] = __builtin_amdgcn_wmma_f32_16x16x32_bf16(
                false, a, false, bl, (short)0, accL[t], false, false);
            accR[t] = __builtin_amdgcn_wmma_f32_16x16x32_bf16(
                false, a, false, br, (short)0, accR[t], false, false);
        }
    }

    // C/D layout: VGPR i -> row mbase + i + 8*(lane>>4), col = t*16 + (lane&15)
    int orow = mbase + (lane >> 4) * 8;
    #pragma unroll
    for (int t = 0; t < 4; ++t) {
        #pragma unroll
        for (int i = 0; i < 8; ++i) {
            int r = orow + i;
            if (r < N) {
                YL[(long long)r * HID + t * 16 + col] = accL[t][i];
                YR[(long long)r * HID + t * 16 + col] = accR[t][i];
            }
        }
    }
}

// Edge scatter: agg[dst[e]][f] += Y[src[e]][f]. One thread per (edge, feature).
// A wave covers 32 contiguous features of one edge -> coalesced 128B load + atomics.
__global__ void scatter_add(const float* __restrict__ Y,
                            const int* __restrict__ src,
                            const int* __restrict__ dst,
                            float* __restrict__ agg, int E) {
    long long t = (long long)blockIdx.x * blockDim.x + threadIdx.x;
    if (t >= (long long)E * HID) return;
    int e = (int)(t >> 6);
    int f = (int)(t & 63);
    int s = src[e], d = dst[e];
    atomicAdd(&agg[(long long)d * HID + f], Y[(long long)s * HID + f]);
}

// out = (agg / max(cnt,1)) + bias + yr   (optionally ReLU)
__global__ void finish_layer(const float* __restrict__ agg,
                             const float* __restrict__ cnt,
                             const float* __restrict__ bias,
                             const float* __restrict__ yr,
                             float* __restrict__ out, int N, int relu) {
    long long t = (long long)blockIdx.x * blockDim.x + threadIdx.x;
    if (t >= (long long)N * HID) return;
    int n = (int)(t >> 6);
    int f = (int)(t & 63);
    float c = cnt[n];
    c = c > 1.0f ? c : 1.0f;
    float v = agg[t] / c + bias[f] + yr[t];
    if (relu) v = v > 0.0f ? v : 0.0f;
    out[t] = v;
}

// Link scores: out[e] = dot(Z[src[e]], Z[dst[e]]) over 64 features, one wave per edge.
__global__ void decode_edges(const float* __restrict__ Z,
                             const int* __restrict__ src,
                             const int* __restrict__ dst,
                             float* __restrict__ out, int E) {
    int wid  = (int)((blockIdx.x * blockDim.x + threadIdx.x) >> 5);
    int lane = threadIdx.x & 31;
    if (wid >= E) return;
    const float* zs = Z + (long long)src[wid] * HID;
    const float* zd = Z + (long long)dst[wid] * HID;
    float p = zs[lane] * zd[lane] + zs[lane + 32] * zd[lane + 32];
    #pragma unroll
    for (int off = 16; off > 0; off >>= 1)
        p += __shfl_xor(p, off, 32);
    if (lane == 0) out[wid] = p;
}

extern "C" void kernel_launch(void* const* d_in, const int* in_sizes, int n_in,
                              void* d_out, int out_size, void* d_ws, size_t ws_size,
                              hipStream_t stream) {
    const float* x   = (const float*)d_in[0];
    const int*   ei  = (const int*)  d_in[1];
    const float* W1l = (const float*)d_in[2];
    const float* b1  = (const float*)d_in[3];
    const float* W1r = (const float*)d_in[4];
    const float* W2l = (const float*)d_in[5];
    const float* b2  = (const float*)d_in[6];
    const float* W2r = (const float*)d_in[7];
    float* out = (float*)d_out;

    const int IN_CH = 128;
    int N = in_sizes[0] / IN_CH;
    int E = in_sizes[1] / 2;
    const int* src = ei;
    const int* dst = ei + E;

    // Workspace layout (256B-aligned)
    size_t nodeF = (size_t)N * HID * sizeof(float);
    size_t a     = (nodeF + 255) & ~(size_t)255;
    size_t cntB  = (((size_t)N * sizeof(float)) + 255) & ~(size_t)255;
    size_t w1B   = (((size_t)IN_CH * HID * sizeof(__bf16)) + 255) & ~(size_t)255;
    size_t w2B   = (((size_t)HID * HID * sizeof(__bf16)) + 255) & ~(size_t)255;
    char* ws = (char*)d_ws;
    float*  YL   = (float*)(ws);                         // lin_l projection; reused as Z
    float*  YR   = (float*)(ws + a);                     // lin_r projection
    float*  AGG  = (float*)(ws + 2 * a);                 // scatter accumulator
    float*  H    = (float*)(ws + 3 * a);                 // layer-1 hidden
    float*  CNT  = (float*)(ws + 4 * a);                 // in-degree counts
    __bf16* W1lp = (__bf16*)(ws + 4 * a + cntB);
    __bf16* W1rp = (__bf16*)(ws + 4 * a + cntB + w1B);
    __bf16* W2lp = (__bf16*)(ws + 4 * a + cntB + 2 * w1B);
    __bf16* W2rp = (__bf16*)(ws + 4 * a + cntB + 2 * w1B + w2B);

    const int TPB = 256;
    long long nf = (long long)N * HID;
    int blkNF   = (int)((nf + TPB - 1) / TPB);
    int blkN    = (N + TPB - 1) / TPB;
    int blkE    = (E + TPB - 1) / TPB;
    long long ef = (long long)E * HID;
    int blkEF   = (int)((ef + TPB - 1) / TPB);
    int waves   = (N + 15) / 16;
    int blkGEMM = (waves * 32 + TPB - 1) / TPB;
    long long ew = (long long)E * 32;
    int blkDEC  = (int)((ew + TPB - 1) / TPB);
    int frag1   = (IN_CH / 32) * 4 * 32;                 // fragments for K=128
    int frag2   = (HID   / 32) * 4 * 32;                 // fragments for K=64

    // ---- pack weights into WMMA B-fragment order (tiny, once per launch) ----
    pack_weight<<<(frag1 + TPB - 1) / TPB, TPB, 0, stream>>>(W1l, W1lp, IN_CH);
    pack_weight<<<(frag1 + TPB - 1) / TPB, TPB, 0, stream>>>(W1r, W1rp, IN_CH);
    pack_weight<<<(frag2 + TPB - 1) / TPB, TPB, 0, stream>>>(W2l, W2lp, HID);
    pack_weight<<<(frag2 + TPB - 1) / TPB, TPB, 0, stream>>>(W2r, W2rp, HID);

    // ---- degree counts (shared by both layers) ----
    zero_f32<<<blkN, TPB, 0, stream>>>(CNT, (long long)N);
    count_edges<<<blkE, TPB, 0, stream>>>(dst, CNT, E);

    // ---- layer 1: project first (128->64), then aggregate 64-wide ----
    zero_f32<<<blkNF, TPB, 0, stream>>>(AGG, nf);
    gemm_dual<<<blkGEMM, TPB, 0, stream>>>(x, W1lp, W1rp, YL, YR, N, IN_CH);
    scatter_add<<<blkEF, TPB, 0, stream>>>(YL, src, dst, AGG, E);
    finish_layer<<<blkNF, TPB, 0, stream>>>(AGG, CNT, b1, YR, H, N, 1);

    // ---- layer 2 ----
    zero_f32<<<blkNF, TPB, 0, stream>>>(AGG, nf);
    gemm_dual<<<blkGEMM, TPB, 0, stream>>>(H, W2lp, W2rp, YL, YR, N, HID);
    scatter_add<<<blkEF, TPB, 0, stream>>>(YL, src, dst, AGG, E);
    finish_layer<<<blkNF, TPB, 0, stream>>>(AGG, CNT, b2, YR, YL, N, 0); // Z -> YL

    // ---- decode: per-edge dot products ----
    decode_edges<<<blkDEC, TPB, 0, stream>>>(YL, src, dst, out, E);
}
